// LoRA_42374147342669
// MI455X (gfx1250) — matrix operand order repaired and checked by reading, not compiled, not run on hardware
//
#include <hip/hip_runtime.h>

// ---------------------------------------------------------------------------
// MoRA/LoRA-routing kernel for MI455X (gfx1250, wave32, WMMA).
// out = x @ W^T + b + (masked (x@A^T)) @ B^T, mask per-token from gate argmax.
// bf16 WMMA (16x16x32) with f32 accumulation: the problem is compute-bound
// (AI ~800 FLOP/B vs machine balance ~100), and bf16 WMMA has 8x the K-depth
// per issue of the f32 16x16x4 path. Double-buffered LDS + software pipeline:
// global loads for tile t+1 are in flight while tile t's WMMAs execute.
// ---------------------------------------------------------------------------

typedef __bf16 bf16_t;
typedef __attribute__((ext_vector_type(16))) __bf16 v16bf;
typedef __attribute__((ext_vector_type(8)))  __bf16 v8bf;
typedef __attribute__((ext_vector_type(4)))  __bf16 v4bf;
typedef __attribute__((ext_vector_type(8)))  float  v8f;
typedef __attribute__((ext_vector_type(4)))  float  v4f;

#define D_IN   4096
#define D_OUT  4096
#define NTOK   8192      // 4 * 2048
#define TA_LD  96        // ta workspace row stride: 64 lora cols + 4 gate + pad

// ---- helpers ---------------------------------------------------------------

__device__ __forceinline__ v16bf cat8(v8bf a, v8bf b) {
    return __builtin_shufflevector(a, b, 0,1,2,3,4,5,6,7,8,9,10,11,12,13,14,15);
}

__device__ __forceinline__ v4bf cvtr(v4f f) {
    v4bf r = { (__bf16)f.x, (__bf16)f.y, (__bf16)f.z, (__bf16)f.w };
    return r;
}

// A-fragment (16x32 bf16, M x K): lane m%16; lanes 0-15 take K{0..7,16..23},
// lanes 16-31 take K{8..15,24..31}. tile is row-major [rows][32].
__device__ __forceinline__ v16bf frag_a(const bf16_t* tile, int row, int half) {
    const bf16_t* p = tile + row * 32;
    v8bf lo = *(const v8bf*)(p + half * 8);
    v8bf hi = *(const v8bf*)(p + 16 + half * 8);
    return cat8(lo, hi);
}

// B-fragment (32x16 bf16, K x N): lane holds column N=lane%16; lanes 0-15 take
// K{0..7,16..23}, lanes 16-31 take K{8..15,24..31}. tile is [cols][32] (K-contiguous).
__device__ __forceinline__ v16bf frag_b(const bf16_t* tile, int col, int half) {
    const bf16_t* p = tile + col * 32;
    v8bf lo = *(const v8bf*)(p + half * 8);
    v8bf hi = *(const v8bf*)(p + 16 + half * 8);
    return cat8(lo, hi);
}

// ---------------------------------------------------------------------------
// Kernel 1: ta_raw[8192][96] = x @ [A(64) ; gate_w(4) ; zeros(28)]^T  (f32 out)
// BM=128, BN=96, BK=32; 256 thr = 8 waves as 4(M) x 2(N); wave tile 32x48.
// Branch-free staging: B-tile vecs 0-1 -> A rows [0,64); vec 2 -> gate rows
// (row = 64 + tid/8; rows >= 68 read a clamped gate row scaled by 0).
// ---------------------------------------------------------------------------
__global__ __launch_bounds__(256)
void lora_ta_gate_gemm(const float* __restrict__ hs,     // [NTOK][D_IN]
                       const float* __restrict__ Alora,  // [64][D_IN]
                       const float* __restrict__ gw,     // [4][D_IN]
                       float* __restrict__ ta)           // [NTOK][TA_LD]
{
    __shared__ bf16_t Asm[2][128 * 32];   // 16 KB
    __shared__ bf16_t Bsm[2][96 * 32];    // 12 KB

    const int tid  = threadIdx.x;
    const int lane = tid & 31;
    const int wave = tid >> 5;
    const int wm   = wave & 3;    // 0..3 -> 32-row strip
    const int wn   = wave >> 2;   // 0..1 -> 48-col strip
    const int row0 = blockIdx.x * 128;
    const int l16  = lane & 15;
    const int lh   = lane >> 4;

    // third B-staging vector: gate region, rows 64..95 (>=68 are zero pad)
    const int   e2     = (512 + tid) * 4;
    const int   r2     = e2 >> 5;                       // 64..95
    const int   c2     = e2 & 31;
    const int   grow   = (r2 - 64) < 3 ? (r2 - 64) : 3; // clamp (gw has 4 rows)
    const float gscale = (r2 < 68) ? 1.0f : 0.0f;

    v8f acc[2][3];
    for (int i = 0; i < 2; ++i)
        for (int j = 0; j < 3; ++j)
            for (int e = 0; e < 8; ++e) acc[i][j][e] = 0.0f;

    v4f ar[4], br[3];
    auto load_g = [&](int kt) {
        for (int i = 0; i < 4; ++i) {
            int e = (i * 256 + tid) * 4;
            int r = e >> 5, c = e & 31;
            ar[i] = *(const v4f*)(hs + (size_t)(row0 + r) * D_IN + kt + c);
        }
        for (int i = 0; i < 2; ++i) {
            int e = (i * 256 + tid) * 4;
            int r = e >> 5, c = e & 31;
            br[i] = *(const v4f*)(Alora + (size_t)r * D_IN + kt + c);
        }
        v4f g = *(const v4f*)(gw + (size_t)grow * D_IN + kt + c2);
        br[2] = g * gscale;
    };
    auto store_s = [&](int buf) {
        for (int i = 0; i < 4; ++i)
            *(v4bf*)&Asm[buf][(i * 256 + tid) * 4] = cvtr(ar[i]);
        for (int i = 0; i < 3; ++i)
            *(v4bf*)&Bsm[buf][(i * 256 + tid) * 4] = cvtr(br[i]);
    };
    auto compute = [&](int buf) {
        v16bf af[2], bfr[3];
        for (int ms = 0; ms < 2; ++ms) af[ms]  = frag_a(&Asm[buf][0], wm * 32 + ms * 16 + l16, lh);
        for (int ns = 0; ns < 3; ++ns) bfr[ns] = frag_b(&Bsm[buf][0], wn * 48 + ns * 16 + l16, lh);
        for (int ms = 0; ms < 2; ++ms)
            for (int ns = 0; ns < 3; ++ns)
                acc[ms][ns] = __builtin_amdgcn_wmma_f32_16x16x32_bf16(
                    false, af[ms], false, bfr[ns], (short)0, acc[ms][ns], false, false);
    };

    // software pipeline: one barrier per K-step, global loads overlap WMMAs
    load_g(0);
    store_s(0);
    __syncthreads();
    const int T = D_IN / 32;   // 128
    for (int t = 0; t < T; ++t) {
        int cur = t & 1;
        if (t + 1 < T) load_g((t + 1) * 32);
        compute(cur);
        if (t + 1 < T) {
            store_s(cur ^ 1);
            __syncthreads();
        }
    }

    // epilogue: C layout -> lane col = l16, rows lh*8 + v
    for (int ms = 0; ms < 2; ++ms)
        for (int ns = 0; ns < 3; ++ns) {
            int col   = wn * 48 + ns * 16 + l16;
            int rbase = row0 + wm * 32 + ms * 16 + lh * 8;
            for (int v = 0; v < 8; ++v)
                ta[(size_t)(rbase + v) * TA_LD + col] = acc[ms][ns][v];
        }
}

// ---------------------------------------------------------------------------
// Kernel 2: per-token gate argmax (softmax monotone, topk=1 => weight==1),
// zero ta[n][j] for j >= RANKS[argmax]. In place: writes cols 0-63 only.
// ---------------------------------------------------------------------------
__global__ __launch_bounds__(256)
void lora_route_mask(float* __restrict__ ta, const float* __restrict__ gb)
{
    int g = blockIdx.x * 256 + threadIdx.x;   // g = n*64 + j
    int n = g >> 6;
    int j = g & 63;
    const float* row = ta + (size_t)n * TA_LD;
    float best = row[64] + gb[0];
    int bi = 0;
    for (int k = 1; k < 4; ++k) {
        float s = row[64 + k] + gb[k];
        if (s > best) { best = s; bi = k; }   // strict > : first-max tie-break
    }
    const int ranks[4] = {8, 16, 32, 64};
    int r = ranks[bi];
    float v = (j < r) ? row[j] : 0.0f;
    ta[(size_t)n * TA_LD + j] = v;
}

// ---------------------------------------------------------------------------
// Kernel 3: out = x @ W^T + b + ta_masked @ B^T. Single accumulation chain:
// 130 K-steps; the last two switch operands to (ta, Bmat) (uniform select).
// BM=BN=128, BK=32; 8 waves as 2(M) x 4(N); wave tile 64x32 -> 4x2 subtiles.
// Double-buffered LDS + register staging pipeline.
// ---------------------------------------------------------------------------
__global__ __launch_bounds__(256)
void lora_main_gemm(const float* __restrict__ hs,    // [NTOK][D_IN]
                    const float* __restrict__ W,     // [D_OUT][D_IN]
                    const float* __restrict__ bias,  // [D_OUT]
                    const float* __restrict__ Bmat,  // [D_OUT][64]
                    const float* __restrict__ ta,    // [NTOK][TA_LD] masked
                    float* __restrict__ out)         // [NTOK][D_OUT]
{
    __shared__ bf16_t Asm[2][128 * 32];   // 16 KB
    __shared__ bf16_t Bsm[2][128 * 32];   // 16 KB

    const int tid  = threadIdx.x;
    const int lane = tid & 31;
    const int wave = tid >> 5;
    const int wm   = wave >> 2;   // 0..1 -> 64-row strip
    const int wn   = wave & 3;    // 0..3 -> 32-col strip
    const int row0 = blockIdx.x * 128;
    const int col0 = blockIdx.y * 128;
    const int l16  = lane & 15;
    const int lh   = lane >> 4;

    v8f acc[4][2];
    for (int i = 0; i < 4; ++i)
        for (int j = 0; j < 2; ++j)
            for (int e = 0; e < 8; ++e) acc[i][j][e] = 0.0f;

    v4f ar[4], br[4];
    auto load_g = [&](int t) {
        const bool lorap = (t >= D_IN / 32);      // uniform across block
        const int  kt    = t * 32;
        const int  kk    = lorap ? (kt - D_IN) : kt;
        for (int i = 0; i < 4; ++i) {
            int e = (i * 256 + tid) * 4;
            int r = e >> 5, c = e & 31;
            const float* sa = lorap ? (ta   + (size_t)(row0 + r) * TA_LD + kk + c)
                                    : (hs   + (size_t)(row0 + r) * D_IN  + kt + c);
            const float* sb = lorap ? (Bmat + (size_t)(col0 + r) * 64    + kk + c)
                                    : (W    + (size_t)(col0 + r) * D_IN  + kt + c);
            ar[i] = *(const v4f*)sa;
            br[i] = *(const v4f*)sb;
        }
    };
    auto store_s = [&](int buf) {
        for (int i = 0; i < 4; ++i) {
            int e = (i * 256 + tid) * 4;
            *(v4bf*)&Asm[buf][e] = cvtr(ar[i]);
            *(v4bf*)&Bsm[buf][e] = cvtr(br[i]);
        }
    };
    auto compute = [&](int buf) {
        v16bf af[4], bfr[2];
        for (int ms = 0; ms < 4; ++ms) af[ms]  = frag_a(&Asm[buf][0], wm * 64 + ms * 16 + l16, lh);
        for (int ns = 0; ns < 2; ++ns) bfr[ns] = frag_b(&Bsm[buf][0], wn * 32 + ns * 16 + l16, lh);
        for (int ms = 0; ms < 4; ++ms)
            for (int ns = 0; ns < 2; ++ns)
                acc[ms][ns] = __builtin_amdgcn_wmma_f32_16x16x32_bf16(
                    false, af[ms], false, bfr[ns], (short)0, acc[ms][ns], false, false);
    };

    load_g(0);
    store_s(0);
    __syncthreads();
    const int T = (D_IN + 64) / 32;   // 130 = 128 main + 2 lora tail
    for (int t = 0; t < T; ++t) {
        int cur = t & 1;
        if (t + 1 < T) load_g(t + 1);
        compute(cur);
        if (t + 1 < T) {
            store_s(cur ^ 1);
            __syncthreads();
        }
    }

    for (int ms = 0; ms < 4; ++ms)
        for (int ns = 0; ns < 2; ++ns) {
            int col   = col0 + wn * 32 + ns * 16 + l16;
            float bv  = bias[col];
            int rbase = row0 + wm * 64 + ms * 16 + lh * 8;
            for (int v = 0; v < 8; ++v)
                out[(size_t)(rbase + v) * D_OUT + col] = acc[ms][ns][v] + bv;
        }
}

// ---------------------------------------------------------------------------

extern "C" void kernel_launch(void* const* d_in, const int* in_sizes, int n_in,
                              void* d_out, int out_size, void* d_ws, size_t ws_size,
                              hipStream_t stream)
{
    (void)in_sizes; (void)n_in; (void)out_size; (void)ws_size;
    const float* hs = (const float*)d_in[0];  // hidden_states [4,2048,4096]
    const float* W  = (const float*)d_in[1];  // [4096,4096]
    const float* b  = (const float*)d_in[2];  // [4096]
    const float* A  = (const float*)d_in[3];  // [64,4096]
    const float* B  = (const float*)d_in[4];  // [4096,64]
    const float* gw = (const float*)d_in[5];  // [4,4096]
    const float* gb = (const float*)d_in[6];  // [4]
    float* out = (float*)d_out;
    float* ta  = (float*)d_ws;                // [8192][96] f32 (~3 MB)

    lora_ta_gate_gemm<<<dim3(NTOK / 128), 256, 0, stream>>>(hs, A, gw, ta);
    lora_route_mask<<<dim3(NTOK * 64 / 256), 256, 0, stream>>>(ta, gb);
    lora_main_gemm<<<dim3(NTOK / 128, D_OUT / 128), 256, 0, stream>>>(hs, W, b, B, ta, out);
}
